// CompressedGNN_57062935495086
// MI455X (gfx1250) — matrix operand order repaired and checked by reading, not compiled, not run on hardware
//
#include <hip/hip_runtime.h>
#include <hip/hip_bf16.h>

typedef __attribute__((ext_vector_type(16))) _Float16 v16h;
typedef __attribute__((ext_vector_type(8)))  _Float16 v8h;
typedef __attribute__((ext_vector_type(8)))  float    v8f;

#define KDIM 512          // feature dim (both layers); compile-time for LDS sizing
#define BPAD 520          // padded LDS row (halves): 1040B ≡ 4 banks mod 64 → conflict-free

// ---------------------------------------------------------------------------
// Elementwise helpers
// ---------------------------------------------------------------------------

__global__ void k_fill_f32(float* __restrict__ p, float v, int n) {
    int i = blockIdx.x * blockDim.x + threadIdx.x;
    if (i < n) p[i] = v;
}

__global__ void k_deg_scatter(const long long* __restrict__ dst,
                              float* __restrict__ deg, int E) {
    int i = blockIdx.x * blockDim.x + threadIdx.x;
    if (i < E) atomicAdd(&deg[(int)dst[i]], 1.0f);
}

__global__ void k_rsqrt_inplace(float* __restrict__ p, int n) {
    int i = blockIdx.x * blockDim.x + threadIdx.x;
    if (i < n) p[i] = rsqrtf(p[i]);
}

// f32 -> f16 conversion with optional fused relu
__global__ void k_cvt_f16(const float* __restrict__ in,
                          _Float16* __restrict__ out, int n, int doRelu) {
    int i = blockIdx.x * blockDim.x + threadIdx.x;
    if (i < n) {
        float v = in[i];
        if (doRelu) v = fmaxf(v, 0.0f);
        out[i] = (_Float16)v;
    }
}

// W [K x N] f32 row-major  ->  Wt [N x K] f16 row-major (K contiguous per row)
__global__ void k_transpose_w_f16(const float* __restrict__ W,
                                  _Float16* __restrict__ Wt, int K, int N) {
    int i = blockIdx.x * blockDim.x + threadIdx.x;
    if (i < K * N) {
        int k = i / N, nn = i - k * N;
        Wt[(size_t)nn * K + k] = (_Float16)W[i];
    }
}

// ---------------------------------------------------------------------------
// WMMA GEMM: C[M,N] = A[M,K](f16) * Bt[N,K](f16, row = output col) (+bias)
// Grid: (ceil(Mtiles/8), Ntiles). One block = 8 waves = 8 consecutive M-tiles
// sharing one 16-column weight tile. The weight tile is staged into LDS once
// per block via GLOBAL_LOAD_ASYNC_TO_LDS_B128 (ASYNCcnt), then each wave runs
// K/32 v_wmma_f32_16x16x32_f16 ops, A fragments streamed from global (L2-
// resident), B fragments read bank-conflict-free from LDS.
// ---------------------------------------------------------------------------
__global__ __launch_bounds__(256) void k_gemm_wmma_f16(
    const _Float16* __restrict__ A,    // [M x KDIM]
    const _Float16* __restrict__ Bt,   // [>=Ntiles*16 x KDIM]
    const float*    __restrict__ bias, // [N] or nullptr
    float*          __restrict__ C,    // [M x N]
    int M, int N, int Mtiles)
{
    __shared__ _Float16 sB[16][BPAD];

    const int tile_n = blockIdx.y;
    const int wave   = (int)threadIdx.x >> 5;
    const int tile_m = blockIdx.x * 8 + wave;

    // ---- stage 16 x KDIM f16 weight tile into LDS (async, b128 chunks) ----
    // 16 rows * 1024B = 1024 16B-chunks; 256 threads * 4 chunks each.
    {
        const unsigned ldsBase = (unsigned)(unsigned long long)&sB[0][0];
        const char* gBase = (const char*)(Bt + (size_t)tile_n * 16 * KDIM);
        #pragma unroll
        for (int i = 0; i < 4; ++i) {
            const int c    = i * 256 + (int)threadIdx.x;   // chunk id
            const int row  = c >> 6;                       // 64 chunks per row
            const int boff = (c & 63) * 16;                // byte offset in row
            const unsigned long long ga =
                (unsigned long long)(gBase + (size_t)row * (KDIM * 2) + boff);
            const unsigned la = ldsBase + (unsigned)(row * (BPAD * 2) + boff);
            asm volatile("global_load_async_to_lds_b128 %0, %1, off"
                         :: "v"(la), "v"(ga) : "memory");
        }
        asm volatile("s_wait_asynccnt 0x0" ::: "memory");
    }
    __syncthreads();

    if (tile_m >= Mtiles) return;

    const int lane = (int)threadIdx.x & 31;
    const int r = lane & 15;       // A-row / B-col / C-col within tile
    const int g = lane >> 4;       // lane-group (K-half selector)

    int arowIdx = tile_m * 16 + r;
    if (arowIdx >= M) arowIdx = M - 1;                 // safety clamp
    const _Float16* __restrict__ arow = A + (size_t)arowIdx * KDIM;
    const _Float16* brow = &sB[r][0];

    v8f acc = {};
    for (int k0 = 0; k0 < KDIM; k0 += 32) {
        // A fragment: elems 0..15 <- K=k0+g*16..
        v8h alo = *(const v8h*)(arow + k0 + g * 16);
        v8h ahi = *(const v8h*)(arow + k0 + g * 16 + 8);
        v16h a = __builtin_shufflevector(alo, ahi,
                 0,1,2,3,4,5,6,7,8,9,10,11,12,13,14,15);
        // B fragment: elems 0..15 <- K=k0+g*16.. of column r (LDS, padded)
        v16h b = *(const v16h*)(brow + k0 + g * 16);

        if (k0 + 32 < KDIM)                      // global_prefetch next A chunk
            __builtin_prefetch(arow + k0 + 32, 0, 0);

        acc = __builtin_amdgcn_wmma_f32_16x16x32_f16(
                  false, a, false, b, (short)0, acc, false, false);
    }

    // C layout: lane col = r; VGPR v holds row v + g*8
    const int col = tile_n * 16 + r;
    if (col < N) {
        const float bv = bias ? bias[col] : 0.0f;
        #pragma unroll
        for (int v = 0; v < 8; ++v) {
            const int m = tile_m * 16 + v + g * 8;
            if (m < M) C[(size_t)m * N + col] = acc[v] + bv;
        }
    }
}

// ---------------------------------------------------------------------------
// Edge scatter: agg[dst] += h[src] * dinv[src]*dinv[dst]
// one thread = (edge, 4 contiguous features)
// ---------------------------------------------------------------------------
__global__ __launch_bounds__(256) void k_edge_scatter(
    const long long* __restrict__ src, const long long* __restrict__ dst,
    const float* __restrict__ dinv, const float* __restrict__ h,
    float* __restrict__ agg, int E, int F)
{
    const long long t = (long long)blockIdx.x * blockDim.x + threadIdx.x;
    const int chunks = F >> 2;
    if (t >= (long long)E * chunks) return;
    const int e  = (int)(t / chunks);
    const int f4 = (int)(t - (long long)e * chunks) * 4;
    const int s = (int)src[e];
    const int d = (int)dst[e];
    const float nm = dinv[s] * dinv[d];
    const float4 hv = *(const float4*)(h + (size_t)s * F + f4);
    float* ap = agg + (size_t)d * F + f4;
    atomicAdd(ap + 0, hv.x * nm);
    atomicAdd(ap + 1, hv.y * nm);
    atomicAdd(ap + 2, hv.z * nm);
    atomicAdd(ap + 3, hv.w * nm);
}

// agg += h * dinv^2 + bias   (self-loop contribution + bias)
__global__ void k_self_bias(const float* __restrict__ h,
                            const float* __restrict__ dinv,
                            const float* __restrict__ b,
                            float* __restrict__ agg, int n, int F)
{
    const long long t = (long long)blockIdx.x * blockDim.x + threadIdx.x;
    if (t >= (long long)n * F) return;
    const int i = (int)(t / F);
    const int f = (int)(t - (long long)i * F);
    const float di = dinv[i];
    agg[t] += h[t] * di * di + b[f];
}

// ---------------------------------------------------------------------------
extern "C" void kernel_launch(void* const* d_in, const int* in_sizes, int n_in,
                              void* d_out, int out_size, void* d_ws, size_t ws_size,
                              hipStream_t stream)
{
    const float*     x  = (const float*)d_in[0];
    const long long* ei = (const long long*)d_in[1];
    const float*     W1 = (const float*)d_in[2];
    const float*     b1 = (const float*)d_in[3];
    const float*     W2 = (const float*)d_in[4];
    const float*     b2 = (const float*)d_in[5];
    const float*     Wc = (const float*)d_in[6];
    const float*     bc = (const float*)d_in[7];
    float* out = (float*)d_out;

    const int K  = KDIM;
    const int n  = in_sizes[0] / K;   // 10000 nodes
    const int E  = in_sizes[1] / 2;   // 160000 edges
    const int NC = in_sizes[7];       // 100 classes
    const long long* srcIdx = ei;
    const long long* dstIdx = ei + E;

    // Workspace layout (256B aligned slices)
    char* ws = (char*)d_ws;
    size_t off = 0;
    auto alloc = [&](size_t bytes) -> char* {
        char* p = ws + off;
        off = (off + bytes + 255) & ~(size_t)255;
        return p;
    };
    float*    dinv   = (float*)   alloc((size_t)n * sizeof(float));
    float*    hbuf   = (float*)   alloc((size_t)n * K * sizeof(float));
    float*    aggbuf = (float*)   alloc((size_t)n * K * sizeof(float));
    _Float16* in16   = (_Float16*)alloc((size_t)n * K * sizeof(_Float16));
    _Float16* w16    = (_Float16*)alloc((size_t)K * K * sizeof(_Float16));
    (void)ws_size;

    const int BS = 256;
    auto blocks = [](long long total, int bs) -> int {
        return (int)((total + bs - 1) / bs);
    };

    const int Mtiles = (n + 15) / 16;     // 625
    const int NtH    = K / 16;            // 32 tiles for hidden dims
    const int NtC    = (NC + 15) / 16;    // 7 tiles for classifier
    const dim3 gemmGridH((Mtiles + 7) / 8, NtH);
    const dim3 gemmGridC((Mtiles + 7) / 8, NtC);
    const long long edgeWork = (long long)E * (K >> 2);
    const long long nodeWork = (long long)n * K;

    // --- degrees & symmetric norm ------------------------------------------
    k_fill_f32     <<<blocks(n, BS), BS, 0, stream>>>(dinv, 1.0f, n);
    k_deg_scatter  <<<blocks(E, BS), BS, 0, stream>>>(dstIdx, dinv, E);
    k_rsqrt_inplace<<<blocks(n, BS), BS, 0, stream>>>(dinv, n);

    // --- layer 1: h1 = gcn(x, W1, b1); relu --------------------------------
    k_cvt_f16        <<<blocks(nodeWork, BS), BS, 0, stream>>>(x, in16, n * K, 0);
    k_transpose_w_f16<<<blocks((long long)K * K, BS), BS, 0, stream>>>(W1, w16, K, K);
    k_gemm_wmma_f16  <<<gemmGridH, BS, 0, stream>>>(
        in16, w16, nullptr, hbuf, n, K, Mtiles);
    k_fill_f32       <<<blocks(nodeWork, BS), BS, 0, stream>>>(aggbuf, 0.0f, n * K);
    k_edge_scatter   <<<blocks(edgeWork, BS), BS, 0, stream>>>(
        srcIdx, dstIdx, dinv, hbuf, aggbuf, E, K);
    k_self_bias      <<<blocks(nodeWork, BS), BS, 0, stream>>>(
        hbuf, dinv, b1, aggbuf, n, K);

    // --- layer 2: h2 = gcn(relu(h1), W2, b2) -------------------------------
    k_cvt_f16        <<<blocks(nodeWork, BS), BS, 0, stream>>>(aggbuf, in16, n * K, 1);
    k_transpose_w_f16<<<blocks((long long)K * K, BS), BS, 0, stream>>>(W2, w16, K, K);
    k_gemm_wmma_f16  <<<gemmGridH, BS, 0, stream>>>(
        in16, w16, nullptr, hbuf, n, K, Mtiles);
    k_fill_f32       <<<blocks(nodeWork, BS), BS, 0, stream>>>(aggbuf, 0.0f, n * K);
    k_edge_scatter   <<<blocks(edgeWork, BS), BS, 0, stream>>>(
        srcIdx, dstIdx, dinv, hbuf, aggbuf, E, K);
    k_self_bias      <<<blocks(nodeWork, BS), BS, 0, stream>>>(
        hbuf, dinv, b2, aggbuf, n, K);

    // --- classifier: out = h2 @ Wc + bc ------------------------------------
    k_cvt_f16        <<<blocks(nodeWork, BS), BS, 0, stream>>>(aggbuf, in16, n * K, 0);
    k_transpose_w_f16<<<blocks((long long)K * NC, BS), BS, 0, stream>>>(Wc, w16, K, NC);
    k_gemm_wmma_f16  <<<gemmGridC, BS, 0, stream>>>(
        in16, w16, bc, out, n, NC, Mtiles);
}
